// MD_LSTM_4440996184850
// MI455X (gfx1250) — matrix operand order repaired and checked by reading, not compiled, not run on hardware
//
#include <hip/hip_runtime.h>
#include <hip/hip_bf16.h>
#include <stdint.h>

// ---------------------------------------------------------------------------
// CDNA5 (gfx1250) wave32 WMMA bf16 LSTM implementation.
//  - bf16 WMMA (v_wmma_f32_16x16x32_bf16) for all GEMMs, f32 accumulate
//  - weights pre-packed into B-tile layout, staged through LDS with
//    global_load_async_to_lds_b128 (ASYNCcnt) and double buffering
//  - branch-free, phase-peeled K loops (x-input phase, h-recurrence phase)
//  - one fused kernel per LSTM layer per timestep (launch = grid-wide sync)
// ---------------------------------------------------------------------------

typedef __attribute__((ext_vector_type(16))) __bf16        v16bf;
typedef __attribute__((ext_vector_type(8)))  float         v8f;
typedef __attribute__((ext_vector_type(4)))  unsigned int  u32x4;
typedef __attribute__((ext_vector_type(8)))  unsigned int  u32x8;

#define BS     256
#define TI     256
#define FD     128
#define HD     1024
#define G4H    4096
#define TSTEPS 512
#define OUTT   (TI + TSTEPS) /* 768 */

__device__ __forceinline__ unsigned short f2bf(float x) {
  unsigned int u = __float_as_uint(x);
  u += 0x7FFFu + ((u >> 16) & 1u);          // round-to-nearest-even
  return (unsigned short)(u >> 16);
}

__device__ __forceinline__ float sigmoidf_(float x) {
  return 1.0f / (1.0f + __expf(-x));
}

// ---- A-matrix fragment: 16x32 bf16, row-major activations -----------------
// ISA layout: lanes 0-15 (M=lane) hold K = [0,8) in v0..3 and [16,24) in v4..7;
//             lanes 16-31 (M=lane-16) hold K = [8,16) and [24,32).
__device__ __forceinline__ v16bf load_a_frag(const unsigned short* __restrict__ A,
                                             size_t lda, int m0, int k0, int lane) {
  const int Ml  = lane & 15;
  const int klo = (lane & 16) ? 8 : 0;
  const unsigned short* p = A + (size_t)(m0 + Ml) * lda + (size_t)(k0 + klo);
  u32x8 w;
  w.lo = *(const u32x4*)(p);        // K = klo .. klo+7
  w.hi = *(const u32x4*)(p + 16);   // K = klo+16 .. klo+23
  return __builtin_bit_cast(v16bf, w);
}

// ---- B-matrix fragment from pre-packed tiles (32B contiguous per lane) ----
__device__ __forceinline__ v16bf load_b_frag_g(const unsigned short* __restrict__ Bp,
                                               int tile, int lane) {
  return *(const v16bf*)(Bp + (size_t)tile * 512 + (size_t)lane * 16);
}

// ---- CDNA5 async global->LDS copy (32B per lane; offset applies to both) --
__device__ __forceinline__ void async_b32bytes(unsigned lds_off,
                                               unsigned long long gaddr) {
  asm volatile("global_load_async_to_lds_b128 %0, %1, off"
               :: "v"(lds_off), "v"(gaddr) : "memory");
  asm volatile("global_load_async_to_lds_b128 %0, %1, off offset:16"
               :: "v"(lds_off), "v"(gaddr) : "memory");
}
__device__ __forceinline__ void wait_async_1() {
  asm volatile("s_wait_asynccnt 0x1" ::: "memory");
}
__device__ __forceinline__ void wait_async_0() {
  asm volatile("s_wait_asynccnt 0x0" ::: "memory");
}

// ---------------------------------------------------------------------------
// Fused LSTM cell layer: z = x@W_ih^T + h@W_hh^T + b ; gate math; c,h update.
// grid = (2 m-groups, 64 n-tiles), block = 128 (4 waves).
// Each wave owns TWO 16-row m-tiles; the block's 4 waves share one 16-wide
// hidden tile and its 4 gate B-tiles, which are staged in LDS (wave == gate,
// 1KB per gate per K-step, double buffered with ASYNCcnt).
// nk1 must be even (4 or 32 at our call sites) so buffer parity per phase
// reduces to kt&1.
// ---------------------------------------------------------------------------
__global__ __launch_bounds__(128)
void lstm_layer_kernel(const unsigned short* __restrict__ A1, int lda1, int nk1,
                       const unsigned short* __restrict__ Bp1,   // packed W_ih
                       const unsigned short* __restrict__ Aprev, // h_{t-1} bf16
                       const unsigned short* __restrict__ Bp2,   // packed W_hh
                       const float* __restrict__ bias,           // b_ih+b_hh (4H)
                       float* __restrict__ cstate,               // f32 [BS x H]
                       unsigned short* __restrict__ h_out)       // bf16 [BS x H]
{
  __shared__ unsigned short smem[2 * 2048];   // 2 x 4KB double buffer

  const int lane  = threadIdx.x & 31;
  const int wave  = threadIdx.x >> 5;         // == gate id for staging
  const int ntile = blockIdx.y;               // 0..63 hidden 16-col tile
  const int m0    = (blockIdx.x * 8 + wave * 2) * 16;
  const int nkh   = HD >> 5;                  // 32

  // Per-thread global source base for this thread's 32B staging slice.
  const unsigned long long b1 =
      (unsigned long long)(uintptr_t)Bp1 +
      (unsigned long long)((size_t)((wave << 6) + ntile) * nk1) * 1024ull +
      (unsigned long long)(lane * 32);
  const unsigned long long b2 =
      (unsigned long long)(uintptr_t)Bp2 +
      (unsigned long long)((size_t)((wave << 6) + ntile) * nkh) * 1024ull +
      (unsigned long long)(lane * 32);
  const unsigned lbase = (unsigned)(uintptr_t)&smem[0] +
                         (unsigned)(wave * 1024 + lane * 32);

  const v8f vz = {0.f,0.f,0.f,0.f,0.f,0.f,0.f,0.f};
  v8f acc[2][4] = {{vz, vz, vz, vz}, {vz, vz, vz, vz}};

  // One fused K-step: A frags from global, B frags from LDS, 8 WMMAs.
  auto gemm_step = [&](const unsigned short* __restrict__ A, size_t lda,
                       int k0, int cur) {
    v16bf a0 = load_a_frag(A, lda, m0,      k0, lane);
    v16bf a1 = load_a_frag(A, lda, m0 + 16, k0, lane);
    const unsigned short* sb = &smem[0] + cur * 2048;
#pragma unroll
    for (int g = 0; g < 4; ++g) {
      v16bf b = *(const v16bf*)(sb + g * 512 + lane * 16);
      acc[0][g] = __builtin_amdgcn_wmma_f32_16x16x32_bf16(false, a0, false, b,
                                                          (short)0, acc[0][g], false, false);
      acc[1][g] = __builtin_amdgcn_wmma_f32_16x16x32_bf16(false, a1, false, b,
                                                          (short)0, acc[1][g], false, false);
    }
  };

  // ---- prologue: stage first x-phase tile into buffer 0 ----
  async_b32bytes(lbase, b1);
  int cur = 0;

  // ---- phase A (x input), branch-free main body ----
  for (int kt = 0; kt + 1 < nk1; ++kt) {
    async_b32bytes(lbase + ((unsigned)(cur ^ 1) << 12),
                   b1 + (unsigned long long)(kt + 1) * 1024ull);
    wait_async_1();
    __syncthreads();
    gemm_step(A1, (size_t)lda1, kt << 5, cur);
    __syncthreads();
    cur ^= 1;
  }
  // phase A last iteration: prefetch first h-phase tile
  async_b32bytes(lbase + ((unsigned)(cur ^ 1) << 12), b2);
  wait_async_1();
  __syncthreads();
  gemm_step(A1, (size_t)lda1, (nk1 - 1) << 5, cur);
  __syncthreads();
  cur ^= 1;

  // ---- phase B (h recurrence), branch-free main body ----
  for (int kt = 0; kt + 1 < nkh; ++kt) {
    async_b32bytes(lbase + ((unsigned)(cur ^ 1) << 12),
                   b2 + (unsigned long long)(kt + 1) * 1024ull);
    wait_async_1();
    __syncthreads();
    gemm_step(Aprev, HD, kt << 5, cur);
    __syncthreads();
    cur ^= 1;
  }
  // phase B last iteration: nothing left to prefetch
  wait_async_0();
  __syncthreads();
  gemm_step(Aprev, HD, (nkh - 1) << 5, cur);

  // Epilogue: C/D layout — VGPR r: M = r + 8*(lane>=16), N = lane&15.
  const int N    = lane & 15;
  const int mhi  = (lane & 16) ? 8 : 0;
  const int col  = (ntile << 4) + N;
  const float bi = bias[col];
  const float bf = bias[HD + col];
  const float bg = bias[2 * HD + col];
  const float bo = bias[3 * HD + col];
#pragma unroll
  for (int mt = 0; mt < 2; ++mt) {
#pragma unroll
    for (int r = 0; r < 8; ++r) {
      const int    row = m0 + mt * 16 + mhi + r;
      const size_t idx = (size_t)row * HD + col;
      const float iv = sigmoidf_(acc[mt][0][r] + bi);
      const float fv = sigmoidf_(acc[mt][1][r] + bf);
      const float gv = tanhf(acc[mt][2][r] + bg);
      const float ov = sigmoidf_(acc[mt][3][r] + bo);
      const float cn = fv * cstate[idx] + iv * gv;
      cstate[idx] = cn;
      h_out[idx]  = f2bf(ov * tanhf(cn));
    }
  }
}

// ---------------------------------------------------------------------------
// Output step: dx = (h1 @ W_out^T + b_out) * dy_std + dy_mu ; x_new = x + dx.
// grid = (16, 2), block = 128 (4 waves, ntile 0..7 over 128 output features).
// ---------------------------------------------------------------------------
__global__ __launch_bounds__(128)
void out_step_kernel(const unsigned short* __restrict__ h1b,
                     const unsigned short* __restrict__ BpO,
                     const float* __restrict__ b_out,
                     const float* __restrict__ dy_mu,
                     const float* __restrict__ dy_std,
                     const float* __restrict__ x_prev, long long xp_stride,
                     float* __restrict__ x_out,        long long xo_stride,
                     unsigned short* __restrict__ x_cur)
{
  const int lane  = threadIdx.x & 31;
  const int wave  = threadIdx.x >> 5;
  const int m0    = blockIdx.x << 4;
  const int ntile = (blockIdx.y << 2) + wave;   // 0..7
  const v8f vz    = {0.f,0.f,0.f,0.f,0.f,0.f,0.f,0.f};
  v8f acc = vz;

  for (int kt = 0; kt < (HD >> 5); ++kt) {
    v16bf a = load_a_frag(h1b, HD, m0, kt << 5, lane);
    v16bf b = load_b_frag_g(BpO, ntile * (HD >> 5) + kt, lane);
    acc = __builtin_amdgcn_wmma_f32_16x16x32_bf16(false, a, false, b,
                                                  (short)0, acc, false, false);
  }
  const int N   = lane & 15;
  const int mhi = (lane & 16) ? 8 : 0;
  const int col = (ntile << 4) + N;
  const float bo = b_out[col], mu = dy_mu[col], sd = dy_std[col];
#pragma unroll
  for (int r = 0; r < 8; ++r) {
    const int   row = m0 + mhi + r;
    const float dx  = (acc[r] + bo) * sd + mu;
    const float xn  = x_prev[(size_t)row * (size_t)xp_stride + col] + dx;
    x_out[(size_t)row * (size_t)xo_stride + col] = xn;
    x_cur[row * FD + col] = f2bf(xn);
  }
}

// ---------------------------------------------------------------------------
// Prep kernels.
// ---------------------------------------------------------------------------
// Pack W (Nrows x Kdim, row-major f32) into B-tile layout:
// tile(nt,kt): lane = N + 16*h, per-lane K = [8h,8h+8) then [16+8h,24+8h).
__global__ void pack_weight_kernel(const float* __restrict__ W,
                                   unsigned short* __restrict__ Wp,
                                   int Kdim, int Nrows) {
  const int nK    = Kdim >> 5;
  const int total = (Nrows >> 4) * nK * 32;
  for (int tid = blockIdx.x * blockDim.x + threadIdx.x; tid < total;
       tid += gridDim.x * blockDim.x) {
    const int lane = tid & 31;
    const int tile = tid >> 5;
    const int nt   = tile / nK;
    const int kt   = tile - nt * nK;
    const int N    = lane & 15;
    const int klo  = (lane & 16) ? 8 : 0;
    const float* src = W + (size_t)(nt * 16 + N) * Kdim + kt * 32 + klo;
    unsigned short* dst = Wp + (size_t)tile * 512 + lane * 16;
#pragma unroll
    for (int j = 0; j < 16; ++j) dst[j] = f2bf(src[j < 8 ? j : j + 8]);
  }
}

__global__ void vec_add_kernel(const float* __restrict__ a,
                               const float* __restrict__ b,
                               float* __restrict__ o, int n) {
  int i = blockIdx.x * blockDim.x + threadIdx.x;
  if (i < n) o[i] = a[i] + b[i];
}

__global__ void cvt_bf16_kernel(const float* __restrict__ in,
                                unsigned short* __restrict__ out, size_t n) {
  size_t i = (size_t)blockIdx.x * blockDim.x + threadIdx.x;
  if (i < n) out[i] = f2bf(in[i]);
}

__global__ void zero_u32_kernel(unsigned int* __restrict__ p, size_t n) {
  size_t i = (size_t)blockIdx.x * blockDim.x + threadIdx.x;
  if (i < n) p[i] = 0u;
}

// out[b, 0:TI, :] = x_init[b]  (copied as 16B chunks)
__global__ void copy_init_kernel(const float* __restrict__ in,
                                 float* __restrict__ out) {
  const size_t per_b = (size_t)TI * FD / 4;
  const size_t total = (size_t)BS * per_b;
  size_t i = (size_t)blockIdx.x * blockDim.x + threadIdx.x;
  if (i >= total) return;
  const size_t b   = i / per_b;
  const size_t rem = i - b * per_b;
  const u32x4* src = (const u32x4*)in;
  u32x4*       dst = (u32x4*)out;
  dst[b * ((size_t)OUTT * FD / 4) + rem] = src[i];
}

// ---------------------------------------------------------------------------
extern "C" void kernel_launch(void* const* d_in, const int* in_sizes, int n_in,
                              void* d_out, int out_size, void* d_ws, size_t ws_size,
                              hipStream_t stream) {
  (void)in_sizes; (void)n_in; (void)out_size; (void)ws_size;
  const float* x_init = (const float*)d_in[0];
  const float* W_ih0  = (const float*)d_in[1];
  const float* W_hh0  = (const float*)d_in[2];
  const float* b_ih0  = (const float*)d_in[3];
  const float* b_hh0  = (const float*)d_in[4];
  const float* W_ih1  = (const float*)d_in[5];
  const float* W_hh1  = (const float*)d_in[6];
  const float* b_ih1  = (const float*)d_in[7];
  const float* b_hh1  = (const float*)d_in[8];
  const float* W_out  = (const float*)d_in[9];
  const float* b_out  = (const float*)d_in[10];
  const float* dy_mu  = (const float*)d_in[11];
  const float* dy_std = (const float*)d_in[12];
  // d_in[13] = t (device scalar); host loop count must be static -> TSTEPS.
  float* out = (float*)d_out;

  // ---- workspace layout (256B aligned slabs) ----
  uint8_t* ws = (uint8_t*)d_ws;
  size_t off = 0;
  auto alloc = [&](size_t bytes) -> void* {
    void* p = ws + off;
    off += (bytes + 255) & ~(size_t)255;
    return p;
  };
  unsigned short* wp_ih0 = (unsigned short*)alloc((size_t)G4H * FD * 2);
  unsigned short* wp_hh0 = (unsigned short*)alloc((size_t)G4H * HD * 2);
  unsigned short* wp_ih1 = (unsigned short*)alloc((size_t)G4H * HD * 2);
  unsigned short* wp_hh1 = (unsigned short*)alloc((size_t)G4H * HD * 2);
  unsigned short* wp_out = (unsigned short*)alloc((size_t)FD * HD * 2);
  float*          bias0  = (float*)alloc((size_t)G4H * 4);
  float*          bias1  = (float*)alloc((size_t)G4H * 4);
  unsigned short* xinitb = (unsigned short*)alloc((size_t)BS * TI * FD * 2);
  const size_t state_off = off;
  float*          c0     = (float*)alloc((size_t)BS * HD * 4);
  float*          c1     = (float*)alloc((size_t)BS * HD * 4);
  unsigned short* h0b[2] = {(unsigned short*)alloc((size_t)BS * HD * 2),
                            (unsigned short*)alloc((size_t)BS * HD * 2)};
  unsigned short* h1b[2] = {(unsigned short*)alloc((size_t)BS * HD * 2),
                            (unsigned short*)alloc((size_t)BS * HD * 2)};
  unsigned short* x_cur  = (unsigned short*)alloc((size_t)BS * FD * 2);
  const size_t state_bytes = off - state_off;

  auto cdiv = [](size_t a, size_t b) -> unsigned int {
    return (unsigned int)((a + b - 1) / b);
  };

  // ---- prep: pack weights (bf16 B-tiles), fuse biases, convert x_init ----
  pack_weight_kernel<<<cdiv((size_t)(G4H/16)*(FD/32)*32, 256), 256, 0, stream>>>(W_ih0, wp_ih0, FD, G4H);
  pack_weight_kernel<<<cdiv((size_t)(G4H/16)*(HD/32)*32, 256), 256, 0, stream>>>(W_hh0, wp_hh0, HD, G4H);
  pack_weight_kernel<<<cdiv((size_t)(G4H/16)*(HD/32)*32, 256), 256, 0, stream>>>(W_ih1, wp_ih1, HD, G4H);
  pack_weight_kernel<<<cdiv((size_t)(G4H/16)*(HD/32)*32, 256), 256, 0, stream>>>(W_hh1, wp_hh1, HD, G4H);
  pack_weight_kernel<<<cdiv((size_t)(FD/16)*(HD/32)*32, 256), 256, 0, stream>>>(W_out, wp_out, HD, FD);
  vec_add_kernel<<<cdiv(G4H, 256), 256, 0, stream>>>(b_ih0, b_hh0, bias0, G4H);
  vec_add_kernel<<<cdiv(G4H, 256), 256, 0, stream>>>(b_ih1, b_hh1, bias1, G4H);
  cvt_bf16_kernel<<<cdiv((size_t)BS * TI * FD, 256), 256, 0, stream>>>(x_init, xinitb, (size_t)BS * TI * FD);
  zero_u32_kernel<<<cdiv(state_bytes / 4, 256), 256, 0, stream>>>((unsigned int*)(ws + state_off), state_bytes / 4);
  copy_init_kernel<<<cdiv((size_t)BS * TI * FD / 4, 256), 256, 0, stream>>>(x_init, out);

  const dim3 gL(2, 64), gO(16, 2);
  int p = 0;
  auto step = [&](const unsigned short* xin, int lda1) {
    lstm_layer_kernel<<<gL, 128, 0, stream>>>(xin, lda1, FD / 32, wp_ih0,
                                              h0b[p], wp_hh0, bias0, c0, h0b[1 - p]);
    lstm_layer_kernel<<<gL, 128, 0, stream>>>(h0b[1 - p], HD, HD / 32, wp_ih1,
                                              h1b[p], wp_hh1, bias1, c1, h1b[1 - p]);
    p ^= 1;
  };

  // ---- teacher-forced conditioning pass ----
  for (int t = 0; t < TI; ++t) step(xinitb + (size_t)t * FD, TI * FD);

  // ---- x1 = x_init[:, -1] + dx(last top) ----
  out_step_kernel<<<gO, 128, 0, stream>>>(h1b[p], wp_out, b_out, dy_mu, dy_std,
                                          x_init + (size_t)(TI - 1) * FD, (long long)(TI * FD),
                                          out + (size_t)TI * FD, (long long)(OUTT * FD),
                                          x_cur);

  // ---- autoregressive rollout: t-1 further steps ----
  for (int s = 0; s < TSTEPS - 1; ++s) {
    step(x_cur, FD);
    out_step_kernel<<<gO, 128, 0, stream>>>(h1b[p], wp_out, b_out, dy_mu, dy_std,
                                            out + (size_t)(TI + s) * FD, (long long)(OUTT * FD),
                                            out + (size_t)(TI + s + 1) * FD, (long long)(OUTT * FD),
                                            x_cur);
  }
}
